// ConformerBlock_7988639171244
// MI455X (gfx1250) — matrix operand order repaired and checked by reading, not compiled, not run on hardware
//
#include <hip/hip_runtime.h>
#include <hip/hip_bf16.h>
#include <math.h>

typedef __attribute__((ext_vector_type(16))) _Float16 v16h;
typedef __attribute__((ext_vector_type(8)))  _Float16 v8h;
typedef __attribute__((ext_vector_type(8)))  float    v8f;

union AFrag { v16h v; v8h h[2]; };

// ---------------------------------------------------------------------------
// Constants for this problem
// ---------------------------------------------------------------------------
static constexpr int Mrows = 8192;     // B*T
static constexpr int Dm    = 512;
static constexpr int HID   = 1024;
static constexpr int Cc    = 1024;     // conv channels
static constexpr int TT    = 1024;
static constexpr int NH    = 8;
static constexpr int DH    = 64;

// ---------------------------------------------------------------------------
// Async global -> LDS copy of 16 bytes per lane (CDNA5 ASYNCcnt path).
// GVS addressing: SGPR64 base + signed 32-bit VGPR byte offset.
// ---------------------------------------------------------------------------
__device__ __forceinline__ void async_copy16(unsigned lds_off, const void* base, unsigned goff) {
    asm volatile("global_load_async_to_lds_b128 %0, %1, %2"
                 :
                 : "v"(lds_off), "v"(goff), "s"((unsigned long long)(uintptr_t)base)
                 : "memory");
}
__device__ __forceinline__ void async_wait0() {
    asm volatile("s_wait_asynccnt 0" ::: "memory");
}
__device__ __forceinline__ unsigned lds_addr(const void* p) {
    // __shared__ pointers are 32-bit addrspace(3); ptrtoint yields the LDS offset
    return (unsigned)(__UINTPTR_TYPE__)p;
}

// ---------------------------------------------------------------------------
// Weight pack kernels: f32 -> f16, optionally transposed to (N x K) "Bt"
// ---------------------------------------------------------------------------
__global__ void pack_transpose(const float* __restrict__ src, _Float16* __restrict__ dst,
                               int K, int N) {
    // dst[n*K + k] = (f16) src[k*N + n]
    size_t total = (size_t)K * N;
    for (size_t idx = blockIdx.x * blockDim.x + threadIdx.x; idx < total;
         idx += (size_t)gridDim.x * blockDim.x) {
        int n = (int)(idx % N);
        int k = (int)(idx / N);
        dst[(size_t)n * K + k] = (_Float16)src[idx];
    }
}

__global__ void pack_convert(const float* __restrict__ src, _Float16* __restrict__ dst, size_t n) {
    for (size_t idx = blockIdx.x * blockDim.x + threadIdx.x; idx < n;
         idx += (size_t)gridDim.x * blockDim.x) {
        dst[idx] = (_Float16)src[idx];
    }
}

// ---------------------------------------------------------------------------
// Tiled f16 WMMA GEMM:  C(MxN) = A(MxK,f16,row-major) * Bt(NxK,f16)^T + epilogue
// Block tile 128x128, 8 waves (4x2), wave tile 32x64 = 2x4 WMMA 16x16x32 tiles.
// Double-buffered LDS staged with GLOBAL_LOAD_ASYNC_TO_LDS_B128 (ASYNCcnt).
// Epilogues:
//   0: out16 = f16( swish(acc + bias) )
//   1: out32 = res + 0.5*(acc + bias)
//   2: out32 = acc + bias
//   3: out32 = res + acc + bias
//   4: s = res + acc + bias; out32 = s; out16 = f16(s)
// ---------------------------------------------------------------------------
#define GM_BM 128
#define GM_BN 128
#define GM_BK 32
#define GM_LD 48   // padded halfs per row (96B, 16B aligned)

template<int EPI>
__global__ __launch_bounds__(256)
void gemm_f16_wmma(const _Float16* __restrict__ A, const _Float16* __restrict__ Bt,
                   const float* __restrict__ bias, const float* __restrict__ res,
                   float* __restrict__ out32, _Float16* __restrict__ out16,
                   int M, int N, int K)
{
    __shared__ _Float16 As[2][GM_BM * GM_LD];
    __shared__ _Float16 Bs[2][GM_BN * GM_LD];

    const int tid  = threadIdx.x;
    const int wave = tid >> 5;
    const int lane = tid & 31;
    const int l15  = lane & 15;
    const int hi   = lane >> 4;
    const int wm   = wave >> 1;      // 0..3 -> 32 rows each
    const int wn   = wave & 1;       // 0..1 -> 64 cols each
    const int m_blk = blockIdx.y * GM_BM;
    const int n_blk = blockIdx.x * GM_BN;

    // Per-thread staging coords: 2 threads per row, 16 halfs each
    const int srow = tid >> 1;
    const int scol = (tid & 1) * 16;
    const _Float16* Abase = A  + (size_t)m_blk * K;
    const _Float16* Bbase = Bt + (size_t)n_blk * K;

    auto issue = [&](int buf, int k0) {
        unsigned la = lds_addr(&As[buf][srow * GM_LD + scol]);
        unsigned lb = lds_addr(&Bs[buf][srow * GM_LD + scol]);
        unsigned go = (unsigned)(((size_t)srow * K + k0 + scol) * sizeof(_Float16));
        async_copy16(la,      Abase, go);
        async_copy16(la + 16, Abase, go + 16);
        async_copy16(lb,      Bbase, go);
        async_copy16(lb + 16, Bbase, go + 16);
    };

    v8f acc[2][4] = {};

    const int nsteps = K / GM_BK;
    issue(0, 0);
    for (int it = 0; it < nsteps; ++it) {
        const int cur = it & 1;
        async_wait0();          // this wave's loads into buffer `cur` complete
        __syncthreads();        // everyone's loads complete; prev compute done
        if (it + 1 < nsteps) issue(1 - cur, (it + 1) * GM_BK);

        AFrag a[2], b[4];
        #pragma unroll
        for (int i = 0; i < 2; ++i) {
            int m = wm * 32 + i * 16 + l15;
            // A layout: elem0-7 -> K = hi*8.. ; elem8-15 -> K = 16+hi*8..
            a[i].h[0] = *(const v8h*)&As[cur][m * GM_LD + hi * 8];
            a[i].h[1] = *(const v8h*)&As[cur][m * GM_LD + 16 + hi * 8];
        }
        #pragma unroll
        for (int j = 0; j < 4; ++j) {
            int n = wn * 64 + j * 16 + l15;
            // B layout: elem0-7 -> K = hi*8.. ; elem8-15 -> K = 16+hi*8..
            b[j].h[0] = *(const v8h*)&Bs[cur][n * GM_LD + hi * 8];
            b[j].h[1] = *(const v8h*)&Bs[cur][n * GM_LD + 16 + hi * 8];
        }
        #pragma unroll
        for (int i = 0; i < 2; ++i)
            #pragma unroll
            for (int j = 0; j < 4; ++j)
                acc[i][j] = __builtin_amdgcn_wmma_f32_16x16x32_f16(
                    false, a[i].v, false, b[j].v, (short)0, acc[i][j], false, false);
        __syncthreads();        // all waves done reading `cur` before it is refilled
    }

    // Epilogue (C layout: VGPR r -> row r (lanes 0-15) / r+8 (lanes 16-31); col = l15)
    #pragma unroll
    for (int i = 0; i < 2; ++i) {
        #pragma unroll
        for (int j = 0; j < 4; ++j) {
            int n = n_blk + wn * 64 + j * 16 + l15;
            float bn = bias[n];
            #pragma unroll
            for (int r = 0; r < 8; ++r) {
                int m = m_blk + wm * 32 + i * 16 + r + 8 * hi;
                float v = acc[i][j][r] + bn;
                size_t o = (size_t)m * N + n;
                if (EPI == 0) {
                    float s = v / (1.f + __expf(-v));
                    out16[o] = (_Float16)s;
                } else if (EPI == 1) {
                    out32[o] = res[o] + 0.5f * v;
                } else if (EPI == 2) {
                    out32[o] = v;
                } else if (EPI == 3) {
                    out32[o] = res[o] + v;
                } else {
                    float s = res[o] + v;
                    out32[o] = s;
                    out16[o] = (_Float16)s;
                }
            }
        }
    }
}

// ---------------------------------------------------------------------------
// LayerNorm over D=512 per row; writes f16 (write16=1) or f32 (write16=0)
// ---------------------------------------------------------------------------
__global__ __launch_bounds__(256)
void ln_kernel(const float* __restrict__ x, const float* __restrict__ g,
               const float* __restrict__ be, float* __restrict__ o32,
               _Float16* __restrict__ o16, int write16)
{
    __shared__ float rs[256], rq[256];
    const int tid = threadIdx.x;
    size_t base = (size_t)blockIdx.x * Dm;
    float s = 0.f, q = 0.f;
    for (int i = tid; i < Dm; i += 256) { float v = x[base + i]; s += v; q += v * v; }
    rs[tid] = s; rq[tid] = q; __syncthreads();
    for (int o = 128; o > 0; o >>= 1) {
        if (tid < o) { rs[tid] += rs[tid + o]; rq[tid] += rq[tid + o]; }
        __syncthreads();
    }
    float mean = rs[0] * (1.f / Dm);
    float var  = rq[0] * (1.f / Dm) - mean * mean;
    float inv  = rsqrtf(var + 1e-5f);
    for (int i = tid; i < Dm; i += 256) {
        float v = (x[base + i] - mean) * inv * g[i] + be[i];
        if (write16) o16[base + i] = (_Float16)v;
        else         o32[base + i] = v;
    }
}

// ---------------------------------------------------------------------------
// RoPE: q/k f32 (M x 512 each) -> f16 [B,H,T,dh] with rotation; v plain convert
// ---------------------------------------------------------------------------
__global__ void rope_kernel(const float* __restrict__ q32, const float* __restrict__ k32,
                            const float* __restrict__ v32,
                            _Float16* __restrict__ qh, _Float16* __restrict__ kh,
                            _Float16* __restrict__ vh)
{
    size_t total = (size_t)Mrows * NH * (DH / 2);   // 2,097,152
    for (size_t idx = blockIdx.x * blockDim.x + threadIdx.x; idx < total;
         idx += (size_t)gridDim.x * blockDim.x) {
        int i = (int)(idx & 31);
        int h = (int)((idx >> 5) & 7);
        int m = (int)(idx >> 8);
        int b = m >> 10;
        int t = m & 1023;
        float inv = __powf(10000.f, -((2.f * i) / (float)DH));
        float fr = (float)t * inv;
        float cs = __cosf(fr), sn = __sinf(fr);
        size_t row = (size_t)m * Dm + h * DH + 2 * i;
        size_t ob  = (((size_t)(b * NH + h) * TT) + t) * DH + 2 * i;
        float qe = q32[row], qo = q32[row + 1];
        qh[ob]     = (_Float16)(qe * cs - qo * sn);
        qh[ob + 1] = (_Float16)(qo * cs + qe * sn);
        float ke = k32[row], ko = k32[row + 1];
        kh[ob]     = (_Float16)(ke * cs - ko * sn);
        kh[ob + 1] = (_Float16)(ko * cs + ke * sn);
        vh[ob]     = (_Float16)v32[row];
        vh[ob + 1] = (_Float16)v32[row + 1];
    }
}

// ---------------------------------------------------------------------------
// Flash attention with WMMA. grid = (B*H, T/128), 256 threads (8 waves).
// Each wave owns 16 q rows; K/V streamed through LDS in chunks of 128.
// K tile staged with async global->LDS; V transposed through VGPRs.
// ---------------------------------------------------------------------------
#define AT_KLD 72    // Kt row stride (halfs)
#define AT_VLD 136   // Vt row stride (halfs)
#define AT_PLD 48    // P row stride (halfs)

__global__ __launch_bounds__(256)
void attn_kernel(const _Float16* __restrict__ qh, const _Float16* __restrict__ kh,
                 const _Float16* __restrict__ vh, _Float16* __restrict__ ctx,
                 float scale)
{
    __shared__ _Float16 Kt[128 * AT_KLD];          // [s][d]
    __shared__ _Float16 Vt[64 * AT_VLD];           // [d][s]  (transposed)
    __shared__ _Float16 Ps[8 * 16 * AT_PLD];       // per-wave P bounce

    const int bh   = blockIdx.x;
    const int b    = bh >> 3;
    const int h    = bh & 7;
    const int tid  = threadIdx.x;
    const int wave = tid >> 5;
    const int lane = tid & 31;
    const int l15  = lane & 15;
    const int hi   = lane >> 4;
    const int qrow0 = blockIdx.y * 128 + wave * 16;

    // Resident q fragments (one per 32-wide K-step of dh=64)
    const _Float16* qbase = qh + (((size_t)bh * TT) + qrow0) * DH;
    AFrag qa[2];
    #pragma unroll
    for (int kk = 0; kk < 2; ++kk) {
        qa[kk].h[0] = *(const v8h*)(qbase + (size_t)l15 * DH + kk * 32 + hi * 8);
        qa[kk].h[1] = *(const v8h*)(qbase + (size_t)l15 * DH + kk * 32 + 16 + hi * 8);
    }

    v8f acc[4] = {};
    float mrow[8], lrow[8];
    #pragma unroll
    for (int r = 0; r < 8; ++r) { mrow[r] = -1e30f; lrow[r] = 0.f; }

    _Float16* pb = &Ps[wave * 16 * AT_PLD];
    const _Float16* khbase = kh + (size_t)bh * TT * DH;

    for (int sc = 0; sc < TT; sc += 128) {
        __syncthreads();
        // Async-stage K chunk [128][64] -> Kt (4 x b128 per thread)
        {
            int row = tid >> 1;
            int d0  = (tid & 1) * 32;
            unsigned la = lds_addr(&Kt[row * AT_KLD + d0]);
            unsigned go = (unsigned)((((size_t)(sc + row)) * DH + d0) * sizeof(_Float16));
            #pragma unroll
            for (int q = 0; q < 4; ++q)
                async_copy16(la + q * 16, khbase, go + q * 16);
        }
        // Load V chunk transposed -> Vt[d][s] (through VGPRs)
        {
            int row = tid >> 1;
            int d0  = (tid & 1) * 32;
            const _Float16* src = vh + (((size_t)bh * TT) + sc + row) * DH + d0;
            #pragma unroll
            for (int d = 0; d < 32; ++d)
                Vt[(d0 + d) * AT_VLD + row] = src[d];
        }
        async_wait0();
        __syncthreads();

        for (int st = 0; st < 128; st += 32) {
            // Scores: two 16x16 tiles (s cols st..st+15, st+16..st+31)
            v8f c0 = {}, c1 = {};
            #pragma unroll
            for (int kk = 0; kk < 2; ++kk) {
                AFrag b0, b1;
                b0.h[0] = *(const v8h*)&Kt[(st + l15) * AT_KLD + kk * 32 + hi * 8];
                b0.h[1] = *(const v8h*)&Kt[(st + l15) * AT_KLD + kk * 32 + 16 + hi * 8];
                b1.h[0] = *(const v8h*)&Kt[(st + 16 + l15) * AT_KLD + kk * 32 + hi * 8];
                b1.h[1] = *(const v8h*)&Kt[(st + 16 + l15) * AT_KLD + kk * 32 + 16 + hi * 8];
                c0 = __builtin_amdgcn_wmma_f32_16x16x32_f16(false, qa[kk].v, false, b0.v,
                                                            (short)0, c0, false, false);
                c1 = __builtin_amdgcn_wmma_f32_16x16x32_f16(false, qa[kk].v, false, b1.v,
                                                            (short)0, c1, false, false);
            }

            // Online softmax (row r lives in VGPR r across 16 lanes of a half-wave)
            float alpha[8];
            #pragma unroll
            for (int r = 0; r < 8; ++r) {
                float v0 = c0[r] * scale;
                float v1 = c1[r] * scale;
                float mx = fmaxf(v0, v1);
                #pragma unroll
                for (int o = 8; o >= 1; o >>= 1) mx = fmaxf(mx, __shfl_xor(mx, o, 16));
                float mnew = fmaxf(mrow[r], mx);
                alpha[r] = __expf(mrow[r] - mnew);
                float p0 = __expf(v0 - mnew);
                float p1 = __expf(v1 - mnew);
                float rs = p0 + p1;
                #pragma unroll
                for (int o = 8; o >= 1; o >>= 1) rs += __shfl_xor(rs, o, 16);
                lrow[r] = lrow[r] * alpha[r] + rs;
                mrow[r] = mnew;
                // Write P (row = r + 8*hi, cols l15 / 16+l15)
                pb[(r + 8 * hi) * AT_PLD + l15]      = (_Float16)p0;
                pb[(r + 8 * hi) * AT_PLD + 16 + l15] = (_Float16)p1;
            }
            // Rescale accumulators
            #pragma unroll
            for (int nt = 0; nt < 4; ++nt)
                #pragma unroll
                for (int r = 0; r < 8; ++r) acc[nt][r] *= alpha[r];

            asm volatile("s_wait_dscnt 0" ::: "memory");

            // Re-read P as an A fragment (K = 32 s-positions)
            AFrag pa;
            pa.h[0] = *(const v8h*)&pb[l15 * AT_PLD + hi * 8];
            pa.h[1] = *(const v8h*)&pb[l15 * AT_PLD + 16 + hi * 8];

            // ctx += P @ V  (4 N-tiles over dh=64)
            #pragma unroll
            for (int nt = 0; nt < 4; ++nt) {
                AFrag vb;
                vb.h[0] = *(const v8h*)&Vt[(nt * 16 + l15) * AT_VLD + st + hi * 8];
                vb.h[1] = *(const v8h*)&Vt[(nt * 16 + l15) * AT_VLD + st + 16 + hi * 8];
                acc[nt] = __builtin_amdgcn_wmma_f32_16x16x32_f16(false, pa.v, false, vb.v,
                                                                 (short)0, acc[nt], false, false);
            }
        }
    }

    // Final normalize + store ctx f16 as [b, t, h*64 + d]
    #pragma unroll
    for (int r = 0; r < 8; ++r) {
        float inv = 1.f / lrow[r];
        int trow = qrow0 + r + 8 * hi;
        size_t base = ((size_t)(b * TT + trow)) * Dm + h * DH;
        #pragma unroll
        for (int nt = 0; nt < 4; ++nt)
            ctx[base + nt * 16 + l15] = (_Float16)(acc[nt][r] * inv);
    }
}

// ---------------------------------------------------------------------------
// GLU: u = a * sigmoid(gate), g1 is (M x 2048) f32 -> u (M x 1024) f32
// ---------------------------------------------------------------------------
__global__ void glu_kernel(const float* __restrict__ g1, float* __restrict__ u)
{
    size_t total = (size_t)Mrows * Cc;
    for (size_t idx = blockIdx.x * blockDim.x + threadIdx.x; idx < total;
         idx += (size_t)gridDim.x * blockDim.x) {
        size_t m = idx / Cc;
        int    c = (int)(idx % Cc);
        float a    = g1[m * (2 * Cc) + c];
        float gate = g1[m * (2 * Cc) + Cc + c];
        u[idx] = a * (1.f / (1.f + __expf(-gate)));
    }
}

// ---------------------------------------------------------------------------
// Depthwise conv, K=31, SAME padding over T.  u,w: [B,T,C] f32
// ---------------------------------------------------------------------------
__global__ void dwconv_kernel(const float* __restrict__ u, const float* __restrict__ dw,
                              const float* __restrict__ db, float* __restrict__ w)
{
    size_t total = (size_t)Mrows * Cc;
    for (size_t idx = blockIdx.x * blockDim.x + threadIdx.x; idx < total;
         idx += (size_t)gridDim.x * blockDim.x) {
        int    c = (int)(idx % Cc);
        size_t m = idx / Cc;
        int    t = (int)(m & 1023);
        size_t brow = m - t;
        float acc = db[c];
        #pragma unroll 1
        for (int j = 0; j < 31; ++j) {
            int tt = t + j - 15;
            if (tt >= 0 && tt < TT)
                acc += u[(brow + tt) * Cc + c] * dw[(size_t)j * Cc + c];
        }
        w[idx] = acc;
    }
}

// ---------------------------------------------------------------------------
// BatchNorm stats (mean/var per channel over all B*T rows). grid = C blocks.
// ---------------------------------------------------------------------------
__global__ __launch_bounds__(256)
void bnstats_kernel(const float* __restrict__ w, float* __restrict__ stats)
{
    __shared__ float rs[256], rq[256];
    const int c = blockIdx.x;
    const int tid = threadIdx.x;
    float s = 0.f, q = 0.f;
    for (int r = tid; r < Mrows; r += 256) {
        float v = w[(size_t)r * Cc + c];
        s += v; q += v * v;
    }
    rs[tid] = s; rq[tid] = q; __syncthreads();
    for (int o = 128; o > 0; o >>= 1) {
        if (tid < o) { rs[tid] += rs[tid + o]; rq[tid] += rq[tid + o]; }
        __syncthreads();
    }
    if (tid == 0) {
        float m = rs[0] / (float)Mrows;
        stats[c]      = m;
        stats[Cc + c] = rq[0] / (float)Mrows - m * m;
    }
}

__global__ void bnapply_kernel(const float* __restrict__ w, const float* __restrict__ stats,
                               const float* __restrict__ g, const float* __restrict__ be,
                               _Float16* __restrict__ out16)
{
    size_t total = (size_t)Mrows * Cc;
    for (size_t idx = blockIdx.x * blockDim.x + threadIdx.x; idx < total;
         idx += (size_t)gridDim.x * blockDim.x) {
        int c = (int)(idx % Cc);
        float y = (w[idx] - stats[c]) * rsqrtf(stats[Cc + c] + 1e-5f) * g[c] + be[c];
        float sw = y / (1.f + __expf(-y));
        out16[idx] = (_Float16)sw;
    }
}

// ---------------------------------------------------------------------------
// Host launcher
// ---------------------------------------------------------------------------
static constexpr size_t MB = 1ull << 20;
// weights (f16)
static constexpr size_t OFF_W1T   = 0;                      // 1024x512   (1MB)
static constexpr size_t OFF_W2T   = 1 * MB;                 // 512x1024   (1MB)
static constexpr size_t OFF_WQT   = 2 * MB;                 // 512x512    (0.5MB)
static constexpr size_t OFF_WKT   = 2 * MB + 512 * 1024;
static constexpr size_t OFF_WVT   = 3 * MB;
static constexpr size_t OFF_WOT   = 3 * MB + 512 * 1024;
static constexpr size_t OFF_PW1   = 4 * MB;                 // 2048x512   (2MB)
static constexpr size_t OFF_PW2   = 6 * MB;                 // 512x1024   (1MB)
static constexpr size_t OFF_F2W1T = 7 * MB;
static constexpr size_t OFF_F2W2T = 8 * MB;
// activations
static constexpr size_t OFF_XH    = 12 * MB;   // f16 Mx512      (8MB)
static constexpr size_t OFF_HS    = 20 * MB;   // f16 Mx1024     (16MB)  ffn hidden
static constexpr size_t OFF_Y13   = 36 * MB;   // f32 Mx512      (16MB)  y1 then y3
static constexpr size_t OFF_LNH   = 52 * MB;   // f16 Mx512      (8MB)   LN outputs / y3h
static constexpr size_t OFF_BIG   = 60 * MB;   // f32 Mx2048 max (64MB)  q/k/v then g1
static constexpr size_t OFF_QH    = 124 * MB;  // f16 [B,H,T,dh] (8MB)
static constexpr size_t OFF_KH    = 132 * MB;
static constexpr size_t OFF_VH    = 140 * MB;
static constexpr size_t OFF_CTXH  = 148 * MB;  // f16 Mx512      (8MB)
static constexpr size_t OFF_Y24   = 156 * MB;  // f32 Mx512      (16MB)  y2 then y4
static constexpr size_t OFF_U32   = 172 * MB;  // f32 Mx1024     (32MB)
static constexpr size_t OFF_WC    = 204 * MB;  // f32 Mx1024     (32MB)
static constexpr size_t OFF_BNH   = 236 * MB;  // f16 Mx1024     (16MB)
static constexpr size_t OFF_STAT  = 252 * MB;  // f32 2x1024

extern "C" void kernel_launch(void* const* d_in, const int* in_sizes, int n_in,
                              void* d_out, int out_size, void* d_ws, size_t ws_size,
                              hipStream_t stream) {
    const float* x        = (const float*)d_in[0];
    const float* ffn1_w1  = (const float*)d_in[1];
    const float* ffn1_b1  = (const float*)d_in[2];
    const float* ffn1_w2  = (const float*)d_in[3];
    const float* ffn1_b2  = (const float*)d_in[4];
    const float* ln_a_g   = (const float*)d_in[5];
    const float* ln_a_b   = (const float*)d_in[6];
    const float* wq       = (const float*)d_in[7];
    const float* bq       = (const float*)d_in[8];
    const float* wk       = (const float*)d_in[9];
    const float* bk       = (const float*)d_in[10];
    const float* wv       = (const float*)d_in[11];
    const float* bv       = (const float*)d_in[12];
    const float* wo       = (const float*)d_in[13];
    const float* bo       = (const float*)d_in[14];
    const float* c_ln_g   = (const float*)d_in[15];
    const float* c_ln_b   = (const float*)d_in[16];
    const float* pw1_w    = (const float*)d_in[17];
    const float* pw1_b    = (const float*)d_in[18];
    const float* dw_w     = (const float*)d_in[19];
    const float* dw_b     = (const float*)d_in[20];
    const float* bn_g     = (const float*)d_in[21];
    const float* bn_b     = (const float*)d_in[22];
    const float* pw2_w    = (const float*)d_in[23];
    const float* pw2_b    = (const float*)d_in[24];
    const float* ffn2_w1  = (const float*)d_in[25];
    const float* ffn2_b1  = (const float*)d_in[26];
    const float* ffn2_w2  = (const float*)d_in[27];
    const float* ffn2_b2  = (const float*)d_in[28];
    const float* ln_o_g   = (const float*)d_in[29];
    const float* ln_o_b   = (const float*)d_in[30];

    char* ws = (char*)d_ws;
    auto h16 = [&](size_t off) { return (_Float16*)(ws + off); };
    auto f32 = [&](size_t off) { return (float*)(ws + off); };

    const int GS = 2048;  // grid-stride blocks

    // ---- Pack weights ----
    pack_transpose<<<GS, 256, 0, stream>>>(ffn1_w1, h16(OFF_W1T), Dm, HID);
    pack_transpose<<<GS, 256, 0, stream>>>(ffn1_w2, h16(OFF_W2T), HID, Dm);
    pack_transpose<<<GS, 256, 0, stream>>>(wq, h16(OFF_WQT), Dm, Dm);
    pack_transpose<<<GS, 256, 0, stream>>>(wk, h16(OFF_WKT), Dm, Dm);
    pack_transpose<<<GS, 256, 0, stream>>>(wv, h16(OFF_WVT), Dm, Dm);
    pack_transpose<<<GS, 256, 0, stream>>>(wo, h16(OFF_WOT), Dm, Dm);
    pack_convert<<<GS, 256, 0, stream>>>(pw1_w, h16(OFF_PW1), (size_t)2 * Cc * Dm);
    pack_convert<<<GS, 256, 0, stream>>>(pw2_w, h16(OFF_PW2), (size_t)Dm * Cc);
    pack_transpose<<<GS, 256, 0, stream>>>(ffn2_w1, h16(OFF_F2W1T), Dm, HID);
    pack_transpose<<<GS, 256, 0, stream>>>(ffn2_w2, h16(OFF_F2W2T), HID, Dm);
    pack_convert<<<GS, 256, 0, stream>>>(x, h16(OFF_XH), (size_t)Mrows * Dm);

    dim3 blk(256);
    auto ggrid = [&](int N) { return dim3(N / GM_BN, Mrows / GM_BM); };

    // ---- FFN1: y1 = x + 0.5*(swish(x@w1+b1)@w2+b2) ----
    gemm_f16_wmma<0><<<ggrid(HID), blk, 0, stream>>>(h16(OFF_XH), h16(OFF_W1T), ffn1_b1,
        nullptr, nullptr, h16(OFF_HS), Mrows, HID, Dm);
    gemm_f16_wmma<1><<<ggrid(Dm), blk, 0, stream>>>(h16(OFF_HS), h16(OFF_W2T), ffn1_b2,
        x, f32(OFF_Y13), nullptr, Mrows, Dm, HID);

    // ---- MHSA ----
    ln_kernel<<<Mrows, blk, 0, stream>>>(f32(OFF_Y13), ln_a_g, ln_a_b, nullptr, h16(OFF_LNH), 1);
    float* q32 = f32(OFF_BIG);
    float* k32 = f32(OFF_BIG + 16 * MB);
    float* v32 = f32(OFF_BIG + 32 * MB);
    gemm_f16_wmma<2><<<ggrid(Dm), blk, 0, stream>>>(h16(OFF_LNH), h16(OFF_WQT), bq,
        nullptr, q32, nullptr, Mrows, Dm, Dm);
    gemm_f16_wmma<2><<<ggrid(Dm), blk, 0, stream>>>(h16(OFF_LNH), h16(OFF_WKT), bk,
        nullptr, k32, nullptr, Mrows, Dm, Dm);
    gemm_f16_wmma<2><<<ggrid(Dm), blk, 0, stream>>>(h16(OFF_LNH), h16(OFF_WVT), bv,
        nullptr, v32, nullptr, Mrows, Dm, Dm);
    rope_kernel<<<GS, 256, 0, stream>>>(q32, k32, v32, h16(OFF_QH), h16(OFF_KH), h16(OFF_VH));
    attn_kernel<<<dim3(8 * NH, TT / 128), blk, 0, stream>>>(h16(OFF_QH), h16(OFF_KH),
        h16(OFF_VH), h16(OFF_CTXH), 1.f / sqrtf((float)Dm));
    gemm_f16_wmma<3><<<ggrid(Dm), blk, 0, stream>>>(h16(OFF_CTXH), h16(OFF_WOT), bo,
        f32(OFF_Y13), f32(OFF_Y24), nullptr, Mrows, Dm, Dm);

    // ---- Conv module ----
    ln_kernel<<<Mrows, blk, 0, stream>>>(f32(OFF_Y24), c_ln_g, c_ln_b, nullptr, h16(OFF_LNH), 1);
    gemm_f16_wmma<2><<<ggrid(2 * Cc), blk, 0, stream>>>(h16(OFF_LNH), h16(OFF_PW1), pw1_b,
        nullptr, f32(OFF_BIG), nullptr, Mrows, 2 * Cc, Dm);
    glu_kernel<<<GS, 256, 0, stream>>>(f32(OFF_BIG), f32(OFF_U32));
    dwconv_kernel<<<GS, 256, 0, stream>>>(f32(OFF_U32), dw_w, dw_b, f32(OFF_WC));
    bnstats_kernel<<<Cc, blk, 0, stream>>>(f32(OFF_WC), f32(OFF_STAT));
    bnapply_kernel<<<GS, 256, 0, stream>>>(f32(OFF_WC), f32(OFF_STAT), bn_g, bn_b, h16(OFF_BNH));
    gemm_f16_wmma<4><<<ggrid(Dm), blk, 0, stream>>>(h16(OFF_BNH), h16(OFF_PW2), pw2_b,
        f32(OFF_Y24), f32(OFF_Y13), h16(OFF_LNH), Mrows, Dm, Cc);

    // ---- FFN2 ----
    gemm_f16_wmma<0><<<ggrid(HID), blk, 0, stream>>>(h16(OFF_LNH), h16(OFF_F2W1T), ffn2_b1,
        nullptr, nullptr, h16(OFF_HS), Mrows, HID, Dm);
    gemm_f16_wmma<1><<<ggrid(Dm), blk, 0, stream>>>(h16(OFF_HS), h16(OFF_F2W2T), ffn2_b2,
        f32(OFF_Y13), f32(OFF_Y24), nullptr, Mrows, Dm, HID);

    // ---- Final LN -> d_out (f32) ----
    ln_kernel<<<Mrows, blk, 0, stream>>>(f32(OFF_Y24), ln_o_g, ln_o_b, (float*)d_out, nullptr, 0);
}
